// CustomLSTM_3358664426298
// MI455X (gfx1250) — matrix operand order repaired and checked by reading, not compiled, not run on hardware
//
#include <hip/hip_runtime.h>

// ---------------------------------------------------------------------------
// 2-layer LSTM, B=64, T=256, H=512 on gfx1250 (MI455X).
// bf16 WMMA (v_wmma_f32_16x16x32_bf16), fp32 accumulation.
// Weights pre-packed once into WMMA B-fragment order (bf16) so the per-step
// GEMM loop is barrier-free: direct global b128 loads + wmma only.
// One fused kernel per (t, layer): gates GEMM (K=1024) + bias + activations
// + c/h update. Each block owns a 16-col slice of H and all 4 gate types.
// ---------------------------------------------------------------------------

#define B_  64
#define T_  256
#define H_  512
#define G4_ 2048            // 4*H
#define GS_STRIDE 17        // floats per LDS gate row (pad 16 -> 17)

typedef __attribute__((ext_vector_type(16))) __bf16 v16bf;
typedef __attribute__((ext_vector_type(8)))  __bf16 v8bf;
typedef __attribute__((ext_vector_type(8)))  float  v8f;

__device__ __forceinline__ float sigmoidf_(float x) {
    return 1.0f / (1.0f + __expf(-x));
}
__device__ __forceinline__ float tanhf_fast_(float x) {
    // tanh(x) = 2*sigmoid(2x) - 1   (v_exp_f32 fast path, no ocml branches)
    return 2.0f / (1.0f + __expf(-2.0f * x)) - 1.0f;
}

// x [B,T,H] fp32  ->  xb [T,B,H] bf16
__global__ void cvt_x_kernel(const float* __restrict__ x, __bf16* __restrict__ xb) {
    int idx = blockIdx.x * 256 + threadIdx.x;          // over T*B*H, exact grid
    int t   = idx / (B_ * H_);
    int rem = idx - t * (B_ * H_);
    int b   = rem / H_;
    int h   = rem - b * H_;
    xb[idx] = (__bf16)x[(b * T_ + t) * H_ + h];
}

__global__ void zero_ws_kernel(unsigned int* __restrict__ p) {
    p[blockIdx.x * 256 + threadIdx.x] = 0u;            // exact grid
}

// Pack one weight matrix W [512][2048] fp32 into B-fragment order bf16:
//   out[(((ct*16 + kc)*32) + lane)*16 + j] = W[kc*32 + (lane>>4)*16 + j][ct*16 + (lane&15)]
// ct = column tile (0..127), kc = 32-wide K chunk (0..15).
// Contiguous k order (this code's choice, same as A): lane%16 = N, lane/16 selects K half,
// element j -> K = (lane/16)*16 + j.  One 32B block per lane per chunk.
__global__ void pack_w_kernel(const float* __restrict__ W, __bf16* __restrict__ out) {
    int idx  = blockIdx.x * 256 + threadIdx.x;         // over 2^20, exact grid
    int j    = idx & 15;
    int lane = (idx >> 4) & 31;
    int kc   = (idx >> 9) & 15;
    int ct   = idx >> 13;
    int k    = kc * 32 + (lane >> 4) * 16 + j;
    int n    = ct * 16 + (lane & 15);
    out[idx] = (__bf16)W[k * G4_ + n];
}

// One time step of one layer.
// gates[64][2048] = A0 @ Wi + A1 @ Wh + bi + bh  (A* bf16 64x512, W packed bf16)
// block: 512 threads = 16 waves; wave w -> mt = w&3 (M tile), gt = w>>2 (gate).
// block column slice: hcol in [blockIdx.x*16, +16); grid = 32 blocks.
__global__ __launch_bounds__(512) void lstm_step_kernel(
    const __bf16* __restrict__ A0,      // layer input rows [64][512] bf16
    const __bf16* __restrict__ A1,      // h_prev          [64][512] bf16
    const __bf16* __restrict__ Wip,     // packed Wi, 2^20 bf16
    const __bf16* __restrict__ Whp,     // packed Wh, 2^20 bf16
    const float*  __restrict__ bi,      // [2048]
    const float*  __restrict__ bh,      // [2048]
    float*        __restrict__ cst,     // [64][512] fp32, in-place
    __bf16*       __restrict__ hOutB,   // [64][512] bf16 (next-step A)
    float*        __restrict__ hOutF,   // [64][512] fp32 (for hT)
    float*        __restrict__ outs,    // d_out base or nullptr (layer 0)
    int t)
{
    __shared__ float Gs[4 * 64 * GS_STRIDE];           // 17408 B

    const int tid  = threadIdx.x;
    const int lane = tid & 31;
    const int wv   = tid >> 5;
    const int mt   = wv & 3;            // M tile 0..3
    const int gt   = wv >> 2;           // gate type 0..3 (i,f,g,o)
    const int m    = lane & 15;
    const int half = lane >> 4;
    const int nblk = blockIdx.x;        // 16-column slice of H
    const int ct   = gt * 32 + nblk;    // global column tile 0..127

    // A fragment addressing: lane = row (mt*16 + m),
    // contiguous 16-elem run at K offset half*16 (matches packed B).
    const int aoff = (mt * 16 + m) * H_ + half * 16;
    // B fragment: one 32B block per lane per chunk in the packed stream.
    const size_t boff = ((size_t)ct * 16 * 32 + lane) * 16;

    v8f acc0 = {0.f, 0.f, 0.f, 0.f, 0.f, 0.f, 0.f, 0.f};
    v8f acc1 = {0.f, 0.f, 0.f, 0.f, 0.f, 0.f, 0.f, 0.f};

    #pragma unroll 1
    for (int phase = 0; phase < 2; ++phase) {
        const __bf16* Abase = (phase ? A1 : A0) + aoff;
        const __bf16* Bbase = (phase ? Whp : Wip) + boff;
        #pragma unroll 4
        for (int kc = 0; kc < 16; ++kc) {
            const v8bf alo = *(const v8bf*)(Abase + kc * 32);
            const v8bf ahi = *(const v8bf*)(Abase + kc * 32 + 8);
            const v16bf bfv = *(const v16bf*)(Bbase + (size_t)kc * 512);
            v16bf af;
            #pragma unroll
            for (int j = 0; j < 8; ++j) { af[j] = alo[j]; af[j + 8] = ahi[j]; }
            if (kc & 1)
                acc1 = __builtin_amdgcn_wmma_f32_16x16x32_bf16(
                    false, af, false, bfv, (short)0, acc1, false, false);
            else
                acc0 = __builtin_amdgcn_wmma_f32_16x16x32_bf16(
                    false, af, false, bfv, (short)0, acc0, false, false);
        }
    }
    v8f acc = acc0 + acc1;

    // ---- stage gate accumulators to LDS: Gs[gt][row][col] ----
    #pragma unroll
    for (int r = 0; r < 8; ++r) {
        int row = mt * 16 + half * 8 + r;
        Gs[(gt * 64 + row) * GS_STRIDE + m] = acc[r];
    }
    __syncthreads();

    // ---- elementwise: 64 rows x 16 cols, 2 items/thread ----
    #pragma unroll
    for (int it = 0; it < 2; ++it) {
        int item = tid + it * 512;
        int b    = item >> 4;
        int c    = item & 15;
        int hcol = nblk * 16 + c;
        float gi = Gs[(0 * 64 + b) * GS_STRIDE + c] + bi[0 * H_ + hcol] + bh[0 * H_ + hcol];
        float gf = Gs[(1 * 64 + b) * GS_STRIDE + c] + bi[1 * H_ + hcol] + bh[1 * H_ + hcol];
        float gg = Gs[(2 * 64 + b) * GS_STRIDE + c] + bi[2 * H_ + hcol] + bh[2 * H_ + hcol];
        float go = Gs[(3 * 64 + b) * GS_STRIDE + c] + bi[3 * H_ + hcol] + bh[3 * H_ + hcol];
        float i = sigmoidf_(gi);
        float f = sigmoidf_(gf);
        float g = tanhf_fast_(gg);
        float o = sigmoidf_(go);
        int idx = b * H_ + hcol;
        float cn = f * cst[idx] + i * g;
        float hn = o * tanhf_fast_(cn);
        cst[idx]   = cn;
        hOutB[idx] = (__bf16)hn;
        hOutF[idx] = hn;
        if (outs) outs[b * (T_ * H_) + t * H_ + hcol] = hn;
    }
}

// ---------------------------------------------------------------------------
// Workspace layout (bytes):
//   xb      @ 0          : T*B*H bf16           = 16,777,216
//   Wpack   @ 16777216   : 4 x 2^20 bf16        =  8,388,608
//             order: Wi_l0, Wh_l0, Wi_l1, Wh_l1 (2,097,152 B each)
//   h0b[0]  @ 25165824   : B*H bf16 = 65,536
//   h0b[1]  @ 25231360
//   h1b[0]  @ 25296896
//   h1b[1]  @ 25362432
//   hf      @ 25427968   : [2][B][H] fp32 = 262,144
//   cs      @ 25690112   : [2][B][H] fp32 = 262,144
//   total 25,952,256
// ---------------------------------------------------------------------------

extern "C" void kernel_launch(void* const* d_in, const int* in_sizes, int n_in,
                              void* d_out, int out_size, void* d_ws, size_t ws_size,
                              hipStream_t stream) {
    (void)in_sizes; (void)n_in; (void)out_size; (void)ws_size;
    const float* x  = (const float*)d_in[0];
    const float* Wi = (const float*)d_in[1];   // [2][512][2048]
    const float* Wh = (const float*)d_in[2];   // [2][512][2048]
    const float* bi = (const float*)d_in[3];   // [2][2048]
    const float* bh = (const float*)d_in[4];   // [2][2048]
    float* out = (float*)d_out;

    unsigned char* ws = (unsigned char*)d_ws;
    __bf16* xb   = (__bf16*)(ws);
    __bf16* wp[4] = {                           // Wi0, Wh0, Wi1, Wh1
        (__bf16*)(ws + 16777216),
        (__bf16*)(ws + 16777216 + 2097152),
        (__bf16*)(ws + 16777216 + 2 * 2097152),
        (__bf16*)(ws + 16777216 + 3 * 2097152)
    };
    __bf16* h0b[2] = { (__bf16*)(ws + 25165824), (__bf16*)(ws + 25231360) };
    __bf16* h1b[2] = { (__bf16*)(ws + 25296896), (__bf16*)(ws + 25362432) };
    float*  hf  = (float*)(ws + 25427968);     // [2][B][H]
    float*  cs  = (float*)(ws + 25690112);     // [2][B][H]

    const size_t WSTRIDE = (size_t)H_ * G4_;   // fp32 elems per layer matrix

    // x -> bf16 [T,B,H]
    cvt_x_kernel<<<(T_ * B_ * H_) / 256, 256, 0, stream>>>(x, xb);
    // pack weights into WMMA B-fragment order (once per launch)
    pack_w_kernel<<<(1 << 20) / 256, 256, 0, stream>>>(Wi,           wp[0]);
    pack_w_kernel<<<(1 << 20) / 256, 256, 0, stream>>>(Wh,           wp[1]);
    pack_w_kernel<<<(1 << 20) / 256, 256, 0, stream>>>(Wi + WSTRIDE, wp[2]);
    pack_w_kernel<<<(1 << 20) / 256, 256, 0, stream>>>(Wh + WSTRIDE, wp[3]);
    // zero states (h ping-pong + hf + cs): 786432 bytes = 196608 u32
    zero_ws_kernel<<<196608 / 256, 256, 0, stream>>>((unsigned int*)(ws + 25165824));

    for (int t = 0; t < T_; ++t) {
        int cur = t & 1, nxt = (t + 1) & 1;
        // layer 0: input = xb[t], recurrent = h0[cur] -> h0[nxt]
        lstm_step_kernel<<<32, 512, 0, stream>>>(
            xb + (size_t)t * B_ * H_, h0b[cur],
            wp[0], wp[1], bi, bh,
            cs, h0b[nxt], hf, (float*)nullptr, t);
        // layer 1: input = h0[nxt], recurrent = h1[cur] -> h1[nxt]; writes outs
        lstm_step_kernel<<<32, 512, 0, stream>>>(
            h0b[nxt], h1b[cur],
            wp[2], wp[3], bi + G4_, bh + G4_,
            cs + B_ * H_, h1b[nxt], hf + B_ * H_, out, t);
    }

    // tail of d_out: hT [2,B,H] then cT [2,B,H] — contiguous in ws
    hipMemcpyAsync(out + (size_t)B_ * T_ * H_, hf,
                   2 * (size_t)2 * B_ * H_ * sizeof(float),
                   hipMemcpyDeviceToDevice, stream);
}
